// ContactPointMamba_65163243815267
// MI455X (gfx1250) — matrix operand order repaired and checked by reading, not compiled, not run on hardware
//
#include <hip/hip_runtime.h>

typedef __attribute__((ext_vector_type(16))) _Float16 v16h;
typedef __attribute__((ext_vector_type(8)))  _Float16 v8h;
typedef __attribute__((ext_vector_type(8)))  float    v8f;

// ---- problem constants ----
#define CBS 2
#define CN  1024
#define CD  384
#define CDEPTH 12
#define CDI 768
#define CDS 16
#define CDR 24
#define CKC 4
#define CL  2048          // 2*N (concat fwd+bwd sequence)
#define CMT 2048          // BS*N rows (embedding)
#define CMS 4096          // BS*L rows (mamba sequence)
#define CKF 1568          // 1542 padded to multiple of 32

__device__ __forceinline__ float geluf(float x) {
    const float c = 0.7978845608028654f;
    float t = tanhf(c * (x + 0.044715f * x * x * x));
    return 0.5f * x * (1.0f + t);
}
__device__ __forceinline__ float siluf(float x) {
    return x / (1.0f + __expf(-x));
}

__device__ __forceinline__ v16h load_afrag(const _Float16* p) {
    v8h lo = *(const v8h*)(p);
    v8h hi = *(const v8h*)(p + 16);
    v16h a;
    #pragma unroll
    for (int i = 0; i < 8; ++i) { a[i] = lo[i]; a[i + 8] = hi[i]; }
    return a;
}

// =====================================================================
// WMMA GEMM: C[M,N] = A[M,K] (f16, row-major) * Bt[N,K]^T (f16, row-major
// transposed weights) (+ bias) (+ existing C).
// Register-blocked: each wave computes a 32x32 output (2x2 WMMA frags ->
// 4 v_wmma per k-step on 2 A + 2 B fragments).  Block = 8 waves arranged
// 4x2 -> 128x64 block tile.  Requires M%128==0, N%64==0, K%32==0.
// flags: bit0 = add bias[col], bit1 = accumulate into existing C.
// =====================================================================
__global__ __launch_bounds__(256)
void k_gemm_wmma(const _Float16* __restrict__ A, const _Float16* __restrict__ Bt,
                 const float* __restrict__ bias, float* __restrict__ C,
                 int M, int N, int K, int flags) {
    const int lane = threadIdx.x & 31;
    const int wave = threadIdx.x >> 5;
    const int wm = wave & 3;
    const int wn = wave >> 2;
    const int row0 = blockIdx.x * 128 + wm * 32;
    const int col0 = blockIdx.y * 64 + wn * 32;

    // A fragment addressing (ISA 16-bit A 16x32 layout)
    const int rA = row0 + (lane & 15);
    const int kA = (lane >> 4) * 8;       // 0 or 8
    // B fragment addressing (dense 16-bit B 32x16 layout)
    const int cB = col0 + (lane & 15);
    const int kB = (lane >> 4) * 8;       // 0 or 8

    const _Float16* aptr0 = A + (size_t)rA * K + kA;
    const _Float16* aptr1 = aptr0 + (size_t)16 * K;
    const _Float16* bptr0 = Bt + (size_t)cB * K + kB;
    const _Float16* bptr1 = bptr0 + (size_t)16 * K;

    v8f c00 = {}, c01 = {}, c10 = {}, c11 = {};
    for (int k = 0; k < K; k += 32) {
        __builtin_prefetch(aptr0 + k + 256, 0, 1);   // global_prefetch_b8
        __builtin_prefetch(aptr1 + k + 256, 0, 1);
        __builtin_prefetch(bptr0 + k + 256, 0, 1);
        __builtin_prefetch(bptr1 + k + 256, 0, 1);
        v16h a0 = load_afrag(aptr0 + k);
        v16h a1 = load_afrag(aptr1 + k);
        v16h b0 = load_afrag(bptr0 + k);
        v16h b1 = load_afrag(bptr1 + k);
        c00 = __builtin_amdgcn_wmma_f32_16x16x32_f16(false, a0, false, b0, (short)0, c00, false, false);
        c01 = __builtin_amdgcn_wmma_f32_16x16x32_f16(false, a0, false, b1, (short)0, c01, false, false);
        c10 = __builtin_amdgcn_wmma_f32_16x16x32_f16(false, a1, false, b0, (short)0, c10, false, false);
        c11 = __builtin_amdgcn_wmma_f32_16x16x32_f16(false, a1, false, b1, (short)0, c11, false, false);
    }

    // store: C/D layout -> element r: row = rbase + r + 8*(lane>>4), col = cbase + (lane&15)
    const int rowOff = (lane >> 4) * 8;
    const int cl = lane & 15;
    float bc0 = (flags & 1) ? bias[col0 + cl] : 0.0f;
    float bc1 = (flags & 1) ? bias[col0 + 16 + cl] : 0.0f;
    #pragma unroll
    for (int r = 0; r < 8; ++r) {
        int rowa = row0 + rowOff + r;
        int rowb = rowa + 16;
        size_t i00 = (size_t)rowa * N + col0 + cl;
        size_t i01 = (size_t)rowa * N + col0 + 16 + cl;
        size_t i10 = (size_t)rowb * N + col0 + cl;
        size_t i11 = (size_t)rowb * N + col0 + 16 + cl;
        float v00 = c00[r] + bc0, v01 = c01[r] + bc1;
        float v10 = c10[r] + bc0, v11 = c11[r] + bc1;
        if (flags & 2) { v00 += C[i00]; v01 += C[i01]; v10 += C[i10]; v11 += C[i11]; }
        C[i00] = v00; C[i01] = v01; C[i10] = v10; C[i11] = v11;
    }
}

// =====================================================================
// Convert f32 weight W[depth][K][N] -> f16 transposed Wt[depth][Npad][Kpad],
// zero padding.  blockIdx.z = depth.
// =====================================================================
__global__ void k_convert_wt(const float* __restrict__ W, _Float16* __restrict__ dst,
                             int K, int N, int Kpad, int Npad) {
    int i = blockIdx.x * 256 + threadIdx.x;
    if (i >= Npad * Kpad) return;
    const float* Wd = W + (size_t)blockIdx.z * K * N;
    _Float16* Dd = dst + (size_t)blockIdx.z * Npad * Kpad;
    int k = i % Kpad, n = i / Kpad;
    Dd[i] = (k < K && n < N) ? (_Float16)Wd[(size_t)k * N + n] : (_Float16)0.0f;
}

// ---- build fusion matrix (f16, 2048 x 1568, zero padded) ----
__global__ void k_fusion(const float* __restrict__ x, const float* __restrict__ pf,
                         const float* __restrict__ lf, const float* __restrict__ te,
                         const float* __restrict__ xyz, _Float16* __restrict__ dst) {
    int i = blockIdx.x * 256 + threadIdx.x;
    if (i >= CMT * CKF) return;
    int j = i % CKF;
    int p = (i / CKF) % CN;
    int b = i / (CKF * CN);
    float v;
    if (j < 3)          v = x  [((size_t)b * CN + p) * 3 + j];
    else if (j < 515)   v = lf [(size_t)b * 512 + (j - 3)];
    else if (j < 1027)  v = te [(size_t)b * 512 + (j - 515)];
    else if (j < 1539)  v = pf [((size_t)b * CN + p) * 512 + (j - 1027)];
    else if (j < 1542)  v = xyz[((size_t)b * CN + p) * 3 + (j - 1539)];
    else                v = 0.0f;
    dst[i] = (_Float16)v;
}

__global__ void k_posin(const float* __restrict__ xyz, _Float16* __restrict__ dst) {
    int i = blockIdx.x * 256 + threadIdx.x;
    if (i >= CMT * 32) return;
    int j = i & 31, r = i >> 5;
    dst[i] = (j < 3) ? (_Float16)xyz[(size_t)r * 3 + j] : (_Float16)0.0f;
}

// ---- LayerNorm (optionally gelu after), f32 and/or f16 output ----
__global__ __launch_bounds__(128)
void k_layernorm(const float* __restrict__ x, const float* __restrict__ w,
                 const float* __restrict__ b, float* __restrict__ of32,
                 _Float16* __restrict__ of16, int d, int gelu_after) {
    __shared__ float red[128];
    int row = blockIdx.x;
    const float* xr = x + (size_t)row * d;
    int tid = threadIdx.x;
    float acc = 0.0f;
    for (int j = tid; j < d; j += 128) acc += xr[j];
    red[tid] = acc; __syncthreads();
    for (int s = 64; s; s >>= 1) { if (tid < s) red[tid] += red[tid + s]; __syncthreads(); }
    float mean = red[0] / d;
    __syncthreads();
    float va = 0.0f;
    for (int j = tid; j < d; j += 128) { float dv = xr[j] - mean; va += dv * dv; }
    red[tid] = va; __syncthreads();
    for (int s = 64; s; s >>= 1) { if (tid < s) red[tid] += red[tid + s]; __syncthreads(); }
    float inv = rsqrtf(red[0] / d + 1e-5f);
    for (int j = tid; j < d; j += 128) {
        float v = (xr[j] - mean) * inv * w[j] + b[j];
        if (gelu_after) v = geluf(v);
        if (of32) of32[(size_t)row * d + j] = v;
        if (of16) of16[(size_t)row * d + j] = (_Float16)v;
    }
}

__global__ void k_gelu16(const float* __restrict__ in, _Float16* __restrict__ out, int n) {
    int i = blockIdx.x * 256 + threadIdx.x;
    if (i < n) out[i] = (_Float16)geluf(in[i]);
}

// ---- Morton code + stable bitonic argsort (one block per batch/direction) ----
__global__ __launch_bounds__(1024)
void k_morton_sort(const float* __restrict__ xyz, int reversed,
                   int* __restrict__ ord, int* __restrict__ inv) {
    __shared__ unsigned long long key[CN];
    __shared__ float red[CN];
    int b = blockIdx.x, t = threadIdx.x;
    float p[3];
    #pragma unroll
    for (int a = 0; a < 3; ++a) p[a] = xyz[((size_t)b * CN + t) * 3 + a];
    float mn[3], mx[3];
    for (int a = 0; a < 3; ++a) {
        red[t] = p[a]; __syncthreads();
        for (int s = 512; s; s >>= 1) { if (t < s) red[t] = fminf(red[t], red[t + s]); __syncthreads(); }
        mn[a] = red[0]; __syncthreads();
        red[t] = p[a]; __syncthreads();
        for (int s = 512; s; s >>= 1) { if (t < s) red[t] = fmaxf(red[t], red[t + s]); __syncthreads(); }
        mx[a] = red[0]; __syncthreads();
    }
    int g[3];
    #pragma unroll
    for (int a = 0; a < 3; ++a) {
        float v = (p[a] - mn[a]) / (mx[a] - mn[a] + 1e-6f) * 1023.0f;
        v = fminf(fmaxf(v, 0.0f), 1023.0f);
        g[a] = (int)v;
    }
    int gp0 = reversed ? g[2] : g[0];
    int gp1 = g[1];
    int gp2 = reversed ? g[0] : g[2];
    int code = 0;
    for (int bit = 0; bit < 10; ++bit) {
        code |= ((gp0 >> bit) & 1) << (3 * bit + 0);
        code |= ((gp1 >> bit) & 1) << (3 * bit + 1);
        code |= ((gp2 >> bit) & 1) << (3 * bit + 2);
    }
    key[t] = ((unsigned long long)(unsigned)code << 11) | (unsigned)t;  // stable
    __syncthreads();
    for (int ksz = 2; ksz <= CN; ksz <<= 1)
        for (int j = ksz >> 1; j > 0; j >>= 1) {
            int ixj = t ^ j;
            if (ixj > t) {
                unsigned long long a = key[t], bb = key[ixj];
                bool up = ((t & ksz) == 0);
                if ((a > bb) == up) { key[t] = bb; key[ixj] = a; }
            }
            __syncthreads();
        }
    int idx = (int)(key[t] & 2047ULL);
    ord[b * CN + t] = idx;
    inv[b * CN + idx] = t;
}

// ---- hid = gather(h)*g + bb + gather(pos) over both orderings ----
__global__ void k_build_hid(const float* __restrict__ h, const float* __restrict__ pos,
                            const int* __restrict__ ordf, const int* __restrict__ ordb,
                            const float* __restrict__ g1, const float* __restrict__ b1,
                            const float* __restrict__ g2, const float* __restrict__ b2,
                            float* __restrict__ hid) {
    int i = blockIdx.x * 256 + threadIdx.x;
    if (i >= CMS * CD) return;
    int d = i % CD;
    int t = (i / CD) % CL;
    int b = i / (CD * CL);
    int src; const float* g; const float* bb;
    if (t < CN) { src = ordf[b * CN + t];      g = g1; bb = b1; }
    else        { src = ordb[b * CN + t - CN]; g = g2; bb = b2; }
    size_t si = ((size_t)b * CN + src) * CD + d;
    hid[i] = h[si] * g[d] + bb[d] + pos[si];
}

// ---- causal depthwise conv (K=4) + SiLU; writes u (f32) and u16 ----
__global__ void k_conv_silu(const float* __restrict__ xz, const float* __restrict__ cw,
                            const float* __restrict__ cb, float* __restrict__ u,
                            _Float16* __restrict__ u16) {
    int i = blockIdx.x * 256 + threadIdx.x;
    if (i >= CMS * CDI) return;
    int d = i % CDI;
    int l = (i / CDI) % CL;
    int b = i / (CDI * CL);
    float acc = cb[d];
    #pragma unroll
    for (int k = 0; k < CKC; ++k) {
        int ls = l + k - (CKC - 1);
        if (ls >= 0) acc += xz[((size_t)(b * CL + ls)) * (2 * CDI) + d] * cw[d * CKC + k];
    }
    float s = siluf(acc);
    u[i] = s;
    u16[i] = (_Float16)s;
}

// ---- extract dt-rank slice to padded f16 (4096 x 32) ----
__global__ void k_dtl(const float* __restrict__ dbl, _Float16* __restrict__ dtl16) {
    int i = blockIdx.x * 256 + threadIdx.x;
    if (i >= CMS * 32) return;
    int j = i & 31, r = i >> 5;
    dtl16[i] = (j < CDR) ? (_Float16)dbl[(size_t)r * 64 + j] : (_Float16)0.0f;
}

// ---- selective scan: thread per (b,d); B/C staged in LDS per step ----
__global__ __launch_bounds__(256)
void k_scan(const float* __restrict__ dtraw, const float* __restrict__ dtb,
            const float* __restrict__ Alog, const float* __restrict__ Dp,
            const float* __restrict__ dbl, const float* __restrict__ u,
            const float* __restrict__ xz, _Float16* __restrict__ y16) {
    __shared__ float sBC[32];
    const int b = blockIdx.x / (CDI / 256);
    const int d = (blockIdx.x % (CDI / 256)) * 256 + threadIdx.x;
    float Areg[CDS], h[CDS];
    #pragma unroll
    for (int n = 0; n < CDS; ++n) { Areg[n] = -__expf(Alog[d * CDS + n]); h[n] = 0.0f; }
    const float dtbv = dtb[d], Dv = Dp[d];
    for (int l = 0; l < CL; ++l) {
        size_t row = (size_t)b * CL + l;
        if (threadIdx.x < 32) sBC[threadIdx.x] = dbl[row * 64 + CDR + threadIdx.x];
        __syncthreads();
        float dtv = dtraw[row * CDI + d] + dtbv;
        dtv = (dtv > 20.0f) ? dtv : log1pf(__expf(dtv));     // softplus
        float uv = u[row * CDI + d];
        float y = 0.0f;
        #pragma unroll
        for (int n = 0; n < CDS; ++n) {
            float dA = __expf(dtv * Areg[n]);
            h[n] = dA * h[n] + dtv * sBC[n] * uv;
            y += h[n] * sBC[CDS + n];
        }
        y += Dv * uv;
        float zv = xz[row * (2 * CDI) + CDI + d];
        y *= siluf(zv);
        y16[row * CDI + d] = (_Float16)y;
        __syncthreads();
    }
}

// ---- final projection: inverse gathers + (2D -> 3) matmul ----
__global__ void k_out(const float* __restrict__ hidn, const int* __restrict__ invf,
                      const int* __restrict__ invb, const float* __restrict__ outw,
                      const float* __restrict__ outb, float* __restrict__ out) {
    int i = blockIdx.x * 256 + threadIdx.x;
    if (i >= CBS * CN * 3) return;
    int c = i % 3;
    int p = (i / 3) % CN;
    int b = i / (3 * CN);
    int rf = invf[b * CN + p];
    int rb = invb[b * CN + p];
    const float* hf = hidn + ((size_t)b * CL + rf) * CD;
    const float* hb = hidn + ((size_t)b * CL + CN + rb) * CD;
    float acc = outb[c];
    for (int d = 0; d < CD; ++d)
        acc += hf[d] * outw[d * 3 + c] + hb[d] * outw[(CD + d) * 3 + c];
    out[i] = acc;
}

// =====================================================================
extern "C" void kernel_launch(void* const* d_in, const int* in_sizes, int n_in,
                              void* d_out, int out_size, void* d_ws, size_t ws_size,
                              hipStream_t stream) {
    const float* x    = (const float*)d_in[0];
    const float* pf   = (const float*)d_in[1];
    const float* lf   = (const float*)d_in[2];
    const float* te   = (const float*)d_in[3];
    const float* xyz  = (const float*)d_in[4];
    const float* emb_w1   = (const float*)d_in[5];
    const float* emb_b1   = (const float*)d_in[6];
    const float* eln1_w   = (const float*)d_in[7];
    const float* eln1_b   = (const float*)d_in[8];
    const float* emb_w2   = (const float*)d_in[9];
    const float* emb_b2   = (const float*)d_in[10];
    const float* eln2_w   = (const float*)d_in[11];
    const float* eln2_b   = (const float*)d_in[12];
    const float* pos_w1   = (const float*)d_in[13];
    const float* pos_b1   = (const float*)d_in[14];
    const float* pos_w2   = (const float*)d_in[15];
    const float* pos_b2   = (const float*)d_in[16];
    const float* os_g1    = (const float*)d_in[17];
    const float* os_b1    = (const float*)d_in[18];
    const float* os_g2    = (const float*)d_in[19];
    const float* os_b2    = (const float*)d_in[20];
    const float* ln_w     = (const float*)d_in[21];
    const float* ln_b     = (const float*)d_in[22];
    const float* in_proj  = (const float*)d_in[23];
    const float* conv_w   = (const float*)d_in[24];
    const float* conv_b   = (const float*)d_in[25];
    const float* x_proj   = (const float*)d_in[26];
    const float* dt_w     = (const float*)d_in[27];
    const float* dt_b     = (const float*)d_in[28];
    const float* A_log    = (const float*)d_in[29];
    const float* D_p      = (const float*)d_in[30];
    const float* out_proj = (const float*)d_in[31];
    const float* nf_w     = (const float*)d_in[32];
    const float* nf_b     = (const float*)d_in[33];
    const float* out_w    = (const float*)d_in[34];
    const float* out_bias = (const float*)d_in[35];
    float* out = (float*)d_out;

    // ---- bump allocator over workspace ----
    char* ws = (char*)d_ws;
    size_t off = 0;
    auto alloc = [&](size_t bytes) -> void* {
        void* p = ws + off;
        off = (off + bytes + 255) & ~(size_t)255;
        return p;
    };
    _Float16* fusion16 = (_Float16*)alloc((size_t)CMT * CKF * 2);
    _Float16* posin16  = (_Float16*)alloc((size_t)CMT * 32 * 2);
    _Float16* wtE1 = (_Float16*)alloc((size_t)128 * CKF * 2);
    _Float16* wtE2 = (_Float16*)alloc((size_t)CD * 128 * 2);
    _Float16* wtP1 = (_Float16*)alloc((size_t)128 * 32 * 2);
    _Float16* wtP2 = (_Float16*)alloc((size_t)CD * 128 * 2);
    _Float16* wtIP = (_Float16*)alloc((size_t)CDEPTH * 1536 * CD * 2);
    _Float16* wtXP = (_Float16*)alloc((size_t)CDEPTH * 64 * CDI * 2);
    _Float16* wtDT = (_Float16*)alloc((size_t)CDEPTH * CDI * 32 * 2);
    _Float16* wtOP = (_Float16*)alloc((size_t)CDEPTH * CD * CDI * 2);
    float* h1f   = (float*)alloc((size_t)CMT * 128 * 4);
    _Float16* h1_16 = (_Float16*)alloc((size_t)CMT * 128 * 2);
    float* h2f   = (float*)alloc((size_t)CMT * CD * 4);
    float* hbuf  = (float*)alloc((size_t)CMT * CD * 4);
    float* pos1f = (float*)alloc((size_t)CMT * 128 * 4);
    _Float16* pos1_16 = (_Float16*)alloc((size_t)CMT * 128 * 2);
    float* posf  = (float*)alloc((size_t)CMT * CD * 4);
    int* ord_f = (int*)alloc((size_t)CBS * CN * 4);
    int* ord_b = (int*)alloc((size_t)CBS * CN * 4);
    int* inv_f = (int*)alloc((size_t)CBS * CN * 4);
    int* inv_b = (int*)alloc((size_t)CBS * CN * 4);
    float* hid   = (float*)alloc((size_t)CMS * CD * 4);
    _Float16* ln16 = (_Float16*)alloc((size_t)CMS * CD * 2);
    float* xz    = (float*)alloc((size_t)CMS * 1536 * 4);
    float* u     = (float*)alloc((size_t)CMS * CDI * 4);
    _Float16* u16 = (_Float16*)alloc((size_t)CMS * CDI * 2);
    float* dbl   = (float*)alloc((size_t)CMS * 64 * 4);
    _Float16* dtl16 = (_Float16*)alloc((size_t)CMS * 32 * 2);
    float* dtraw = (float*)alloc((size_t)CMS * CDI * 4);
    _Float16* y16 = (_Float16*)alloc((size_t)CMS * CDI * 2);
    float* hidn  = (float*)alloc((size_t)CMS * CD * 4);
    (void)ws_size; (void)in_sizes; (void)n_in; (void)out_size;

    auto cdiv = [](int a, int b) { return (a + b - 1) / b; };

    // ---- convert weights to f16 transposed ----
    k_convert_wt<<<dim3(cdiv(128 * CKF, 256), 1, 1), 256, 0, stream>>>(emb_w1, wtE1, 1542, 128, CKF, 128);
    k_convert_wt<<<dim3(cdiv(CD * 128, 256), 1, 1), 256, 0, stream>>>(emb_w2, wtE2, 128, CD, 128, CD);
    k_convert_wt<<<dim3(cdiv(128 * 32, 256), 1, 1), 256, 0, stream>>>(pos_w1, wtP1, 3, 128, 32, 128);
    k_convert_wt<<<dim3(cdiv(CD * 128, 256), 1, 1), 256, 0, stream>>>(pos_w2, wtP2, 128, CD, 128, CD);
    k_convert_wt<<<dim3(cdiv(1536 * CD, 256), 1, CDEPTH), 256, 0, stream>>>(in_proj,  wtIP, CD, 1536, CD, 1536);
    k_convert_wt<<<dim3(cdiv(64 * CDI, 256), 1, CDEPTH), 256, 0, stream>>>(x_proj,   wtXP, CDI, 56, CDI, 64);
    k_convert_wt<<<dim3(cdiv(CDI * 32, 256), 1, CDEPTH), 256, 0, stream>>>(dt_w,     wtDT, CDR, CDI, 32, CDI);
    k_convert_wt<<<dim3(cdiv(CD * CDI, 256), 1, CDEPTH), 256, 0, stream>>>(out_proj, wtOP, CDI, CD, CDI, CD);

    // ---- embedding path ----
    k_fusion<<<cdiv(CMT * CKF, 256), 256, 0, stream>>>(x, pf, lf, te, xyz, fusion16);
    k_gemm_wmma<<<dim3(CMT / 128, 128 / 64), 256, 0, stream>>>(fusion16, wtE1, emb_b1, h1f, CMT, 128, CKF, 1);
    k_layernorm<<<CMT, 128, 0, stream>>>(h1f, eln1_w, eln1_b, nullptr, h1_16, 128, 1);
    k_gemm_wmma<<<dim3(CMT / 128, CD / 64), 256, 0, stream>>>(h1_16, wtE2, emb_b2, h2f, CMT, CD, 128, 1);
    k_layernorm<<<CMT, 128, 0, stream>>>(h2f, eln2_w, eln2_b, hbuf, nullptr, CD, 0);

    // ---- positional MLP ----
    k_posin<<<cdiv(CMT * 32, 256), 256, 0, stream>>>(xyz, posin16);
    k_gemm_wmma<<<dim3(CMT / 128, 128 / 64), 256, 0, stream>>>(posin16, wtP1, pos_b1, pos1f, CMT, 128, 32, 1);
    k_gelu16<<<cdiv(CMT * 128, 256), 256, 0, stream>>>(pos1f, pos1_16, CMT * 128);
    k_gemm_wmma<<<dim3(CMT / 128, CD / 64), 256, 0, stream>>>(pos1_16, wtP2, pos_b2, posf, CMT, CD, 128, 1);

    // ---- morton sorts (fwd + bwd) ----
    k_morton_sort<<<CBS, CN, 0, stream>>>(xyz, 0, ord_f, inv_f);
    k_morton_sort<<<CBS, CN, 0, stream>>>(xyz, 1, ord_b, inv_b);

    // ---- assemble sequence ----
    k_build_hid<<<cdiv(CMS * CD, 256), 256, 0, stream>>>(hbuf, posf, ord_f, ord_b,
                                                         os_g1, os_b1, os_g2, os_b2, hid);

    // ---- Mamba layers ----
    for (int i = 0; i < CDEPTH; ++i) {
        k_layernorm<<<CMS, 128, 0, stream>>>(hid, ln_w + i * CD, ln_b + i * CD,
                                             nullptr, ln16, CD, 0);
        k_gemm_wmma<<<dim3(CMS / 128, 1536 / 64), 256, 0, stream>>>(
            ln16, wtIP + (size_t)i * 1536 * CD, nullptr, xz, CMS, 1536, CD, 0);
        k_conv_silu<<<cdiv(CMS * CDI, 256), 256, 0, stream>>>(
            xz, conv_w + (size_t)i * CDI * CKC, conv_b + (size_t)i * CDI, u, u16);
        k_gemm_wmma<<<dim3(CMS / 128, 64 / 64), 256, 0, stream>>>(
            u16, wtXP + (size_t)i * 64 * CDI, nullptr, dbl, CMS, 64, CDI, 0);
        k_dtl<<<cdiv(CMS * 32, 256), 256, 0, stream>>>(dbl, dtl16);
        k_gemm_wmma<<<dim3(CMS / 128, CDI / 64), 256, 0, stream>>>(
            dtl16, wtDT + (size_t)i * CDI * 32, nullptr, dtraw, CMS, CDI, 32, 0);
        k_scan<<<CBS * (CDI / 256), 256, 0, stream>>>(
            dtraw, dt_b + (size_t)i * CDI, A_log + (size_t)i * CDI * CDS,
            D_p + (size_t)i * CDI, dbl, u, xz, y16);
        k_gemm_wmma<<<dim3(CMS / 128, CD / 64), 256, 0, stream>>>(
            y16, wtOP + (size_t)i * CD * CDI, nullptr, hid, CMS, CD, CDI, 2 /*accumulate*/);
    }

    // ---- final norm + output projection ----
    k_layernorm<<<CMS, 128, 0, stream>>>(hid, nf_w, nf_b, hidn, nullptr, CD, 0);
    k_out<<<cdiv(CBS * CN * 3, 256), 256, 0, stream>>>(hidn, inv_f, inv_b, out_w, out_bias, out);
}
